// MaskedGNN_86749749444722
// MI455X (gfx1250) — matrix operand order repaired and checked by reading, not compiled, not run on hardware
//
#include <hip/hip_runtime.h>
#include <hip/hip_bf16.h>
#include <math.h>

typedef __attribute__((ext_vector_type(16))) _Float16 v16h;
typedef __attribute__((ext_vector_type(8)))  _Float16 v8h;
typedef __attribute__((ext_vector_type(8)))  float    v8f;

#define Hh   128
#define Nn   64
#define Bb   128
#define Ll   3
#define Ff   7
#define LAT  128
#define NEGV (-1e9f)
#define DCAT 160   // 7 features + 128 pooled + pad -> multiple of 32

struct GnnParams {
  // f32 inputs
  const float *bf, *emb, *mask;
  const float *eb1,*eg1,*eB1,*eb2,*eg2,*eB2;
  const float *bq,*bk,*bv,*be1,*We2,*be2,*bo,*og,*oB;
  const float *db1,*dg1,*dB1,*db2,*dg2,*dB2,*dW3,*db3;
  // pre-transposed f16 weights in workspace: layout [C][K]
  const _Float16 *teW1, *teW2;                 // K=64 / K=128
  const _Float16 *tQ[Ll], *tK[Ll], *tV[Ll];    // K=128
  const _Float16 *tE1a[Ll], *tE1b[Ll], *tO[Ll];// K=128
  const _Float16 *tdW1, *tdW2;                 // K=160 (zero-padded) / K=128
};

// ---------------------------------------------------------------------------
// Weight prep: dst[c*K + k] = (f16) src[k*C + c], zero for k >= kvalid.
// ---------------------------------------------------------------------------
__global__ void __launch_bounds__(256)
prep_wt_kernel(const float* __restrict__ src, _Float16* __restrict__ dst,
               int K, int C, int kvalid)
{
  int idx = blockIdx.x * 256 + threadIdx.x;
  if (idx >= K * C) return;
  int c = idx / K, k = idx - c * K;
  dst[idx] = (k < kvalid) ? (_Float16)src[k * C + c] : (_Float16)0.0f;
}

// ---------------------------------------------------------------------------
// One wave computes a 16-wide column stripe of D = A * B (+bias).
// Ah: f16 row-major (rows = 16*MT); per-lane A fragment = 2 contiguous 16B loads.
// Bt: f16 transposed [C][K];        per-lane B fragment = 2 contiguous 16B loads.
// EPI: 0 = f32 row-major D[row*ldd+col]
//      1 = f16 row-major Dh[row*ldd+col]
//      2 = f16 transposed Dh[col*ldd+row]
// ---------------------------------------------------------------------------
template<int MT, int EPI>
__device__ __forceinline__ void gemm_cols(const _Float16* __restrict__ Ah, int lda,
                                          const _Float16* __restrict__ Bt,
                                          const float* __restrict__ bias,
                                          float* __restrict__ D,
                                          _Float16* __restrict__ Dh, int ldd,
                                          int K, int col0, int lane)
{
  const int g  = lane >> 4;   // K-half select
  const int mn = lane & 15;   // A-row / B-col within tile
  v8f acc[MT];
#pragma unroll
  for (int rt = 0; rt < MT; ++rt) {
#pragma unroll
    for (int r = 0; r < 8; ++r) acc[rt][r] = 0.0f;
  }
  const _Float16* bcol = Bt + (size_t)(col0 + mn) * K;
  for (int kb = 0; kb < K; kb += 32) {
    // B fragment: element i -> K = kb + 16g + i  (16 contiguous halves)
    v8h b0 = *(const v8h*)(bcol + kb + 16 * g);
    v8h b1 = *(const v8h*)(bcol + kb + 16 * g + 8);
    v16h bfr = __builtin_shufflevector(b0, b1, 0,1,2,3,4,5,6,7,8,9,10,11,12,13,14,15);
#pragma unroll
    for (int rt = 0; rt < MT; ++rt) {
      // A fragment: element i -> K = kb + 16g + i  (16 contiguous halves)
      const _Float16* arow = Ah + (rt * 16 + mn) * lda + kb + 16 * g;
      v8h a0 = *(const v8h*)(arow);
      v8h a1 = *(const v8h*)(arow + 8);
      v16h afr = __builtin_shufflevector(a0, a1, 0,1,2,3,4,5,6,7,8,9,10,11,12,13,14,15);
      acc[rt] = __builtin_amdgcn_wmma_f32_16x16x32_f16(
          false, afr, false, bfr, (short)0, acc[rt], false, false);
    }
  }
  const float bv = bias ? bias[col0 + mn] : 0.0f;
#pragma unroll
  for (int rt = 0; rt < MT; ++rt) {
#pragma unroll
    for (int r = 0; r < 8; ++r) {
      int row = rt * 16 + r + 8 * g;       // C/D layout: VGPR r -> M = r + 8g
      float val = acc[rt][r] + bv;
      if constexpr (EPI == 0)      D [row * ldd + col0 + mn] = val;
      else if constexpr (EPI == 1) Dh[row * ldd + col0 + mn] = (_Float16)val;
      else                         Dh[(col0 + mn) * ldd + row] = (_Float16)val;
    }
  }
}

// Row LayerNorm + ReLU (+ optional mask), one row per thread; f32 src, DT dst.
template<typename DT>
__device__ __forceinline__ void ln_relu_row(const float* __restrict__ src,
                                            DT* __restrict__ dst,
                                            const float* __restrict__ gam,
                                            const float* __restrict__ bet,
                                            float maskv, int n)
{
  const float4* s4 = (const float4*)src;
  const int n4 = n >> 2;
  float m = 0.0f;
  for (int h = 0; h < n4; ++h) { float4 a = s4[h]; m += a.x + a.y + a.z + a.w; }
  m /= (float)n;
  float v = 0.0f;
  for (int h = 0; h < n4; ++h) {
    float4 a = s4[h];
    float d0 = a.x - m, d1 = a.y - m, d2 = a.z - m, d3 = a.w - m;
    v += d0 * d0 + d1 * d1 + d2 * d2 + d3 * d3;
  }
  v /= (float)n;
  const float inv = rsqrtf(v + 1e-5f);
  for (int h = 0; h < n4; ++h) {
    float4 a = s4[h];
    dst[4*h+0] = (DT)(fmaxf((a.x - m) * inv * gam[4*h+0] + bet[4*h+0], 0.0f) * maskv);
    dst[4*h+1] = (DT)(fmaxf((a.y - m) * inv * gam[4*h+1] + bet[4*h+1], 0.0f) * maskv);
    dst[4*h+2] = (DT)(fmaxf((a.z - m) * inv * gam[4*h+2] + bet[4*h+2], 0.0f) * maskv);
    dst[4*h+3] = (DT)(fmaxf((a.w - m) * inv * gam[4*h+3] + bet[4*h+3], 0.0f) * maskv);
  }
}

// ---------------------------------------------------------------------------
// Kernel 1: one workgroup (8 wave32) per battery. Encoder + 3 GATv2 layers +
// masked mean pooling; writes concat(features, pooled, 0-pad) f16 [B][160].
// ---------------------------------------------------------------------------
__global__ void __launch_bounds__(256)
gnn_block_kernel(GnnParams p, _Float16* __restrict__ dcatH)
{
  const int b    = blockIdx.x;
  const int tid  = threadIdx.x;
  const int wave = tid >> 5;
  const int lane = tid & 31;
  const int col0 = wave * 16;

  extern __shared__ float smem[];
  float* tmpF = smem;                  // [64][128] f32 pre-LN scratch
  float* hq   = tmpF + Nn * Hh;        // [64][128] f32 (edge-score input)
  float* hk   = hq   + Nn * Hh;        // [64][128] f32
  float* scF  = hk   + Nn * Hh;        // [64][64]  f32 scores
  float* mkb  = scF  + Nn * Nn;        // [64] node mask
  float* be1s = mkb  + Nn;             // [128]
  float* we2s = be1s + Hh;             // [128]
  float* dsh  = we2s + Hh;             // [4] pooling denom (padded)
  _Float16* xsH  = (_Float16*)(dsh + 4); // [64][128] activations f16
  _Float16* tmpH = xsH  + Nn * Hh;       // [64][128] q/k intermediate f16
  _Float16* aggH = tmpH + Nn * Hh;       // [64][128] agg f16 (embH alias at start)
  _Float16* vvT  = aggH + Nn * Hh;       // [128][64] V transposed f16
  _Float16* scH  = vvT  + Hh * Nn;       // [64][64]  attn f16
  _Float16* embH = aggH;                 // [64][64] staging alias

  if (tid < Nn) mkb[tid] = p.mask[b * Nn + tid];

  // stage material embeddings [64][64] as f16
  for (int i = tid; i < Nn * 64; i += 256) embH[i] = (_Float16)p.emb[b * Nn * 64 + i];
  __syncthreads();

  // ---- encoder ----
  gemm_cols<4,0>(embH, 64, p.teW1, p.eb1, tmpF, nullptr, Hh, 64, col0, lane);
  __syncthreads();
  if (tid < Nn) ln_relu_row<_Float16>(tmpF + tid * Hh, xsH + tid * Hh, p.eg1, p.eB1, 1.0f, Hh);
  __syncthreads();
  gemm_cols<4,0>(xsH, Hh, p.teW2, p.eb2, tmpF, nullptr, Hh, Hh, col0, lane);
  __syncthreads();
  if (tid < Nn) ln_relu_row<_Float16>(tmpF + tid * Hh, xsH + tid * Hh, p.eg2, p.eB2, mkb[tid], Hh);
  __syncthreads();

  // ---- GATv2 layers ----
  for (int l = 0; l < Ll; ++l) {
    const float* bq = p.bq + l * Hh;
    const float* bk = p.bk + l * Hh;
    const float* bv = p.bv + l * Hh;
    const float* bo = p.bo + l * Hh;
    const float* og = p.og + l * Hh;
    const float* oB = p.oB + l * Hh;
    const float  be2l = p.be2[l];

    if (tid < Hh) { be1s[tid] = p.be1[l * Hh + tid]; we2s[tid] = p.We2[l * Hh + tid]; }

    // q = x@Wq+bq (f16) ; hq = q@We1[:H] (f32)
    gemm_cols<4,1>(xsH, Hh, p.tQ[l], bq, nullptr, tmpH, Hh, Hh, col0, lane);
    __syncthreads();
    gemm_cols<4,0>(tmpH, Hh, p.tE1a[l], nullptr, hq, nullptr, Hh, Hh, col0, lane);
    __syncthreads();
    // k = x@Wk+bk (f16) ; hk = k@We1[H:] (f32)
    gemm_cols<4,1>(xsH, Hh, p.tK[l], bk, nullptr, tmpH, Hh, Hh, col0, lane);
    __syncthreads();
    gemm_cols<4,0>(tmpH, Hh, p.tE1b[l], nullptr, hk, nullptr, Hh, Hh, col0, lane);
    __syncthreads();
    // v = x@Wv+bv, stored transposed f16 [h][j] for the agg GEMM's B side
    gemm_cols<4,2>(xsH, Hh, p.tV[l], bv, nullptr, vvT, Nn, Hh, col0, lane);
    __syncthreads();

    // scores[i][j] = sum_h relu(hq[i]+hk[j]+be1)*We2 + be2   (float4 LDS reads)
    for (int pr = tid; pr < Nn * Nn; pr += 256) {
      const int i = pr >> 6, j = pr & 63;
      const float4* qi = (const float4*)(hq + i * Hh);
      const float4* kj = (const float4*)(hk + j * Hh);
      const float4* b4 = (const float4*)be1s;
      const float4* w4 = (const float4*)we2s;
      float s = 0.0f;
      for (int h = 0; h < Hh / 4; ++h) {
        float4 a = qi[h], c = kj[h], e = b4[h], w = w4[h];
        s += fmaxf(a.x + c.x + e.x, 0.0f) * w.x;
        s += fmaxf(a.y + c.y + e.y, 0.0f) * w.y;
        s += fmaxf(a.z + c.z + e.z, 0.0f) * w.z;
        s += fmaxf(a.w + c.w + e.w, 0.0f) * w.w;
      }
      scF[pr] = s + be2l;
    }
    __syncthreads();

    // masked softmax per row -> attn (f16) = softmax * mask2d
    if (tid < Nn) {
      const int i = tid;
      const float mi = mkb[i];
      float mx = -3.4e38f;
      for (int j = 0; j < Nn; ++j) {
        float mv = (mi * mkb[j] > 0.0f) ? scF[i * Nn + j] : NEGV;
        mx = fmaxf(mx, mv);
      }
      float sum = 0.0f;
      for (int j = 0; j < Nn; ++j) {
        float mv = (mi * mkb[j] > 0.0f) ? scF[i * Nn + j] : NEGV;
        float e = __expf(mv - mx);
        scF[i * Nn + j] = e;
        sum += e;
      }
      const float rs = 1.0f / sum;
      for (int j = 0; j < Nn; ++j) {
        float m2 = (mi * mkb[j] > 0.0f) ? 1.0f : 0.0f;
        scH[i * Nn + j] = (_Float16)(scF[i * Nn + j] * rs * m2);
      }
    }
    __syncthreads();

    // agg = attn @ v (f16 out) ; A = scH lda=64, K=64 ; B = vvT (LDS f16)
    gemm_cols<4,1>(scH, Nn, vvT, nullptr, nullptr, aggH, Hh, Nn, col0, lane);
    __syncthreads();
    // x = relu(LN(agg@Wo+bo)) * mask
    gemm_cols<4,0>(aggH, Hh, p.tO[l], bo, tmpF, nullptr, Hh, Hh, col0, lane);
    __syncthreads();
    if (tid < Nn) ln_relu_row<_Float16>(tmpF + tid * Hh, xsH + tid * Hh, og, oB, mkb[tid], Hh);
    __syncthreads();
  }

  // ---- masked mean pooling + concat into dcatH[b][160] (f16) ----
  if (tid == 0) {
    float s = 0.0f;
    for (int i = 0; i < Nn; ++i) s += mkb[i];
    dsh[0] = fmaxf(s, 1.0f);
  }
  __syncthreads();
  if (tid < Hh) {
    float s = 0.0f;
    for (int i = 0; i < Nn; ++i) s += (float)xsH[i * Hh + tid];  // already masked
    dcatH[b * DCAT + Ff + tid] = (_Float16)(s / dsh[0]);
  }
  if (tid < Ff) dcatH[b * DCAT + tid] = (_Float16)p.bf[b * Ff + tid];
  if (tid >= Ff + Hh && tid < DCAT) dcatH[b * DCAT + tid] = (_Float16)0.0f;
}

// ---------------------------------------------------------------------------
// Kernel 2: single block decoder over all 128 batteries.
// z = relu(LN(dcat@dW1+db1)); z = relu(LN(z@dW2+db2)); out = z@dW3+db3.
// ---------------------------------------------------------------------------
__global__ void __launch_bounds__(256)
gnn_decoder_kernel(GnnParams p, const _Float16* __restrict__ dcatH,
                   float* __restrict__ out)
{
  const int tid  = threadIdx.x;
  const int wave = tid >> 5;
  const int lane = tid & 31;
  const int col0 = wave * 16;

  extern __shared__ float smem[];
  float*    z  = smem;                       // [128][128] f32 (reused)
  _Float16* zH = (_Float16*)(z + Bb * Hh);   // [128][128] f16

  // tdW1 is zero-padded to K=160 by prep; dcat pad columns are don't-care
  gemm_cols<8,0>(dcatH, DCAT, p.tdW1, p.db1, z, nullptr, Hh, DCAT, col0, lane);
  __syncthreads();
  if (tid < Bb) ln_relu_row<_Float16>(z + tid * Hh, zH + tid * Hh, p.dg1, p.dB1, 1.0f, Hh);
  __syncthreads();

  gemm_cols<8,0>(zH, Hh, p.tdW2, p.db2, z, nullptr, LAT, Hh, col0, lane);
  __syncthreads();
  if (tid < Bb) ln_relu_row<float>(z + tid * LAT, z + tid * LAT, p.dg2, p.dB2, 1.0f, LAT);
  __syncthreads();

  for (int o = tid; o < Bb * Ff; o += 256) {
    const int r = o / Ff, c = o % Ff;
    float s = p.db3[c];
    for (int h = 0; h < LAT; ++h) s += z[r * LAT + h] * p.dW3[h * Ff + c];
    out[o] = s;
  }
}

extern "C" void kernel_launch(void* const* d_in, const int* in_sizes, int n_in,
                              void* d_out, int out_size, void* d_ws, size_t ws_size,
                              hipStream_t stream)
{
  (void)in_sizes; (void)n_in; (void)out_size; (void)ws_size;
  GnnParams p;
  p.bf  = (const float*)d_in[0];
  p.emb = (const float*)d_in[1];
  p.mask= (const float*)d_in[2];
  const float* eW1 = (const float*)d_in[3];  p.eb1 = (const float*)d_in[4];
  p.eg1 = (const float*)d_in[5];  p.eB1 = (const float*)d_in[6];
  const float* eW2 = (const float*)d_in[7];  p.eb2 = (const float*)d_in[8];
  p.eg2 = (const float*)d_in[9];  p.eB2 = (const float*)d_in[10];
  const float* Wq  = (const float*)d_in[11]; p.bq  = (const float*)d_in[12];
  const float* Wk  = (const float*)d_in[13]; p.bk  = (const float*)d_in[14];
  const float* Wv  = (const float*)d_in[15]; p.bv  = (const float*)d_in[16];
  const float* We1 = (const float*)d_in[17]; p.be1 = (const float*)d_in[18];
  p.We2 = (const float*)d_in[19]; p.be2 = (const float*)d_in[20];
  const float* Wo  = (const float*)d_in[21]; p.bo  = (const float*)d_in[22];
  p.og  = (const float*)d_in[23]; p.oB  = (const float*)d_in[24];
  const float* dW1 = (const float*)d_in[25]; p.db1 = (const float*)d_in[26];
  p.dg1 = (const float*)d_in[27]; p.dB1 = (const float*)d_in[28];
  const float* dW2 = (const float*)d_in[29]; p.db2 = (const float*)d_in[30];
  p.dg2 = (const float*)d_in[31]; p.dB2 = (const float*)d_in[32];
  p.dW3 = (const float*)d_in[33]; p.db3 = (const float*)d_in[34];

  // workspace layout: dcat f16 [128][160] (40,960 B), then f16 transposed weights
  _Float16* dcatH = (_Float16*)d_ws;
  _Float16* wt = (_Float16*)((char*)d_ws + (size_t)Bb * DCAT * sizeof(_Float16));
  size_t off = 0;
  auto prep = [&](const float* src, int K, int C, int kvalid) -> const _Float16* {
    _Float16* dst = wt + off;
    off += (size_t)K * C;
    int total = K * C;
    prep_wt_kernel<<<(total + 255) / 256, 256, 0, stream>>>(src, dst, K, C, kvalid);
    return dst;
  };

  p.teW1 = prep(eW1, 64, Hh, 64);
  p.teW2 = prep(eW2, Hh, Hh, Hh);
  for (int l = 0; l < Ll; ++l) {
    p.tQ[l]   = prep(Wq  + (size_t)l * Hh * Hh,             Hh, Hh, Hh);
    p.tK[l]   = prep(Wk  + (size_t)l * Hh * Hh,             Hh, Hh, Hh);
    p.tV[l]   = prep(Wv  + (size_t)l * Hh * Hh,             Hh, Hh, Hh);
    p.tE1a[l] = prep(We1 + (size_t)l * 2 * Hh * Hh,         Hh, Hh, Hh);
    p.tE1b[l] = prep(We1 + (size_t)l * 2 * Hh * Hh + Hh*Hh, Hh, Hh, Hh);
    p.tO[l]   = prep(Wo  + (size_t)l * Hh * Hh,             Hh, Hh, Hh);
  }
  p.tdW1 = prep(dW1, DCAT, Hh, Ff + Hh);   // K padded 135 -> 160 with zeros
  p.tdW2 = prep(dW2, Hh, LAT, Hh);

  // kernel-1 LDS: 3x[64][128] f32 + scores f32 + small arrays + 4.5x[64][128] f16
  const size_t smem1 = (size_t)(3 * Nn * Hh + Nn * Nn + Nn + 2 * Hh + 4) * sizeof(float)
                     + (size_t)(4 * Nn * Hh + Nn * Nn) * sizeof(_Float16);
  const size_t smem2 = (size_t)(Bb * Hh) * sizeof(float)
                     + (size_t)(Bb * Hh) * sizeof(_Float16);

  gnn_block_kernel<<<Bb, 256, smem1, stream>>>(p, dcatH);
  gnn_decoder_kernel<<<1, 256, smem2, stream>>>(p, dcatH, (float*)d_out);
}
